// SimplifiedAttention_5901285064896
// MI455X (gfx1250) — matrix operand order repaired and checked by reading, not compiled, not run on hardware
//
#include <hip/hip_runtime.h>
#include <stdint.h>

// Problem constants (from reference)
#define B_   4
#define C_   256
#define NH_  4
#define HD_  64
#define N_   4096
#define BH_  (B_*NH_)          // 16
#define SCALE_ 0.125f          // 64^-0.5

typedef __attribute__((ext_vector_type(16))) _Float16 v16h;
typedef __attribute__((ext_vector_type(8)))  _Float16 v8h;
typedef __attribute__((ext_vector_type(8)))  float    v8f;
typedef __attribute__((ext_vector_type(4)))  unsigned int v4u;
typedef __attribute__((ext_vector_type(4)))  int      v4i;
typedef __attribute__((ext_vector_type(8)))  int      v8i;

// --- TDM availability / arity detection --------------------------------------
#if defined(__has_builtin)
# if __has_builtin(__builtin_amdgcn_tensor_load_to_lds)
#  define HAVE_TDM 1
# endif
#endif
#if defined(__has_include)
# if __has_include(<hip/amd_detail/amd_gfx1250_TDM.h>)
#  define TDM_ARITY6 1   // amdgpu-toolchain (clang-23/therock) 6-arg builtin
# endif
#endif

#if defined(HAVE_TDM)
// 1-D contiguous TDM load: 16384 fp16 elements (32 KB) global -> LDS.
// D# group0: count=1 | lds_addr | global_addr[56:0] | type=2
// D# group1: data_size=1(2B); tensor_dim0=tile_dim0=16384; tensor_dim1=tile_dim1=1;
//            tensor_dim0_stride=16384; no padding/iterate/multicast.
__device__ __forceinline__ void tdm_load_chunk(unsigned lds_off, const _Float16* gsrc) {
  unsigned long long ga = (unsigned long long)(uintptr_t)gsrc;
  v4u g0 = { 1u,                                   // count=1 (valid user D#)
             lds_off,                              // LDS byte address
             (unsigned)ga,                         // global_addr[31:0]
             (unsigned)((ga >> 32) & 0x01ffffffu) | (2u << 30) }; // [56:32] | type=2
  v8i g1 = { 0x00010000,            // wg_mask=0, data_size=1 (2B)
             (int)(16384u << 16),   // tensor_dim0[15:0] in bits[63:48]
             0x00010000,            // tensor_dim0 hi=0, tensor_dim1=1
             (int)(16384u << 16),   // tile_dim0=16384 in bits[127:112]
             1,                     // tile_dim1=1, tile_dim2=0
             16384,                 // tensor_dim0_stride[31:0]
             0, 0 };
  v4i z4 = { 0, 0, 0, 0 };
#if defined(TDM_ARITY6)
  v8i z8 = { 0, 0, 0, 0, 0, 0, 0, 0 };
  __builtin_amdgcn_tensor_load_to_lds(g0, g1, z4, z4, z8, 0);
#else
  __builtin_amdgcn_tensor_load_to_lds(g0, g1, z4, z4, 0);
#endif
}
#endif

// ---------------------------------------------------------------------------
// WMMA fragment loaders (wave32, 16x16x32 f16 layouts per CDNA5 ISA 7.12.2)
__device__ __forceinline__ v16h frag_a(const _Float16* p, int lane) {
  const int sel = (lane & 16) ? 8 : 0;
  v8h lo = *(const v8h*)(p + sel);
  v8h hi = *(const v8h*)(p + 16 + sel);
  v16h r;
#pragma unroll
  for (int i = 0; i < 8; ++i) { r[i] = lo[i]; r[i + 8] = hi[i]; }
  return r;
}
__device__ __forceinline__ v16h frag_b(const _Float16* p, int lane) {
  const int sel = (lane & 16) ? 8 : 0;
  v8h lo = *(const v8h*)(p + sel);
  v8h hi = *(const v8h*)(p + 16 + sel);
  v16h r;
#pragma unroll
  for (int i = 0; i < 8; ++i) { r[i] = lo[i]; r[i + 8] = hi[i]; }
  return r;
}

// ---------------------------------------------------------------------------
// 1) fp32 -> fp16 weight conversion (keeps [o][c] layout, c contiguous)
__global__ void k_prep_w(const float* __restrict__ wq, const float* __restrict__ wk,
                         _Float16* __restrict__ wq16, _Float16* __restrict__ wk16) {
  int i = blockIdx.x * 256 + threadIdx.x;            // 65536 elements
  wq16[i] = (_Float16)wq[i];
  wk16[i] = (_Float16)wk[i];
}

// ---------------------------------------------------------------------------
// 2) x [b][c][n] fp32 -> xT16 [b][n][c] fp16 (tiled LDS transpose)
__global__ void k_transpose_x(const float* __restrict__ x, _Float16* __restrict__ xT) {
  __shared__ _Float16 t[64][65];
  int b  = blockIdx.z;
  int c0 = blockIdx.y * 64;
  int n0 = blockIdx.x * 64;
  int col  = threadIdx.x & 63;
  int row4 = threadIdx.x >> 6;                       // 0..3
  const float* xp = x + ((size_t)b * C_ + c0) * N_ + n0;
#pragma unroll
  for (int rr = row4; rr < 64; rr += 4)
    t[rr][col] = (_Float16)xp[(size_t)rr * N_ + col];
  __syncthreads();
  _Float16* dp = xT + ((size_t)b * N_ + n0) * C_ + c0;
#pragma unroll
  for (int rr = row4; rr < 64; rr += 4)
    dp[(size_t)rr * C_ + col] = t[col][rr];          // [n][c], c coalesced
}

// ---------------------------------------------------------------------------
// 3a) v[b*C+c] = mean_n x[b][c][n]   (one wave per row)
__global__ void k_vmean(const float* __restrict__ x, float* __restrict__ v) {
  int row  = blockIdx.x * 8 + (threadIdx.x >> 5);    // 1024 rows
  int lane = threadIdx.x & 31;
  const float* p = x + (size_t)row * N_;
  float s = 0.f;
  for (int j = lane; j < N_; j += 32) s += p[j];
#pragma unroll
  for (int off = 16; off; off >>= 1) s += __shfl_xor(s, off, 32);
  if (lane == 0) v[row] = s * (1.0f / N_);
}

// 3b) pv[b][o] = sum_c w_proj[o][c] * v[b][c]
__global__ void k_pv(const float* __restrict__ wproj, const float* __restrict__ v,
                     float* __restrict__ pv) {
  __shared__ float vs[C_];
  int b = blockIdx.x;
  int o = threadIdx.x;
  vs[o] = v[b * C_ + o];
  __syncthreads();
  float s = 0.f;
  const float* wr = wproj + (size_t)o * C_;
  for (int c = 0; c < C_; ++c) s += wr[c] * vs[c];
  pv[b * C_ + o] = s;
}

// ---------------------------------------------------------------------------
// 4) q/k projection GEMM: M=o(256), N=n(4096), K=c(256)
__global__ void k_qkproj(const _Float16* __restrict__ xT,
                         const _Float16* __restrict__ wq16,
                         const _Float16* __restrict__ wk16,
                         _Float16* __restrict__ qT, _Float16* __restrict__ kT) {
  int wid  = blockIdx.x * 8 + (threadIdx.x >> 5);    // 16384 wave-tiles
  int lane = threadIdx.x & 31;
  int nt = wid & 255;
  int ot = (wid >> 8) & 15;
  int b  = wid >> 12;
  int n0 = nt * 16, o0 = ot * 16;
  int lm = lane & 15;

  const _Float16* brow  = xT   + ((size_t)b * N_ + n0 + lm) * C_;
  const _Float16* aqrow = wq16 + (size_t)(o0 + lm) * C_;
  const _Float16* akrow = wk16 + (size_t)(o0 + lm) * C_;

  v8f accq, acck;
#pragma unroll
  for (int r = 0; r < 8; ++r) { accq[r] = 0.f; acck[r] = 0.f; }

#pragma unroll
  for (int kc = 0; kc < C_; kc += 32) {
    v16h bf = frag_b(brow  + kc, lane);
    v16h aq = frag_a(aqrow + kc, lane);
    v16h ak = frag_a(akrow + kc, lane);
    accq = __builtin_amdgcn_wmma_f32_16x16x32_f16(false, aq, false, bf, (short)0, accq, false, false);
    acck = __builtin_amdgcn_wmma_f32_16x16x32_f16(false, ak, false, bf, (short)0, acck, false, false);
  }

  int n     = n0 + lm;
  int obase = o0 + ((lane & 16) ? 8 : 0);
  int h  = obase >> 6;
  int d0 = obase & 63;
  size_t dst = (((size_t)(b * NH_ + h)) * N_ + n) * HD_ + d0;
  v8h hq, hk;
#pragma unroll
  for (int r = 0; r < 8; ++r) { hq[r] = (_Float16)accq[r]; hk[r] = (_Float16)acck[r]; }
  *(v8h*)(qT + dst) = hq;
  *(v8h*)(kT + dst) = hk;
}

// ---------------------------------------------------------------------------
// 5) Score kernel with double-buffered TDM staging of kT chunks.
#define CHUNK_ 256
__global__ void k_scores(const _Float16* __restrict__ qT,
                         const _Float16* __restrict__ kT,
                         float* __restrict__ attnmax) {
  __shared__ _Float16 lds[2][CHUNK_ * HD_];          // 2 x 32 KB = 64 KB
  int bh   = blockIdx.x >> 5;                        // 16 (b,h) pairs
  int nblk = blockIdx.x & 31;                        // 32 row blocks of 128
  int lane = threadIdx.x & 31;
  int lm   = lane & 15;
  int wv   = threadIdx.x >> 5;                       // 8 waves
  int n0   = nblk * 128 + wv * 16;
  // wave-uniform leadership flag behind a genuinely scalar branch
  const bool lead = (__builtin_amdgcn_readfirstlane((int)(threadIdx.x >> 5)) == 0);

  // A fragments (Q rows), kept in registers for the whole sweep
  const _Float16* qrow = qT + ((size_t)bh * N_ + n0 + lm) * HD_;
  v16h a0 = frag_a(qrow,      lane);                 // d = 0..31
  v16h a1 = frag_a(qrow + 32, lane);                 // d = 32..63

  v8f rmax;
#pragma unroll
  for (int r = 0; r < 8; ++r) rmax[r] = -3.4e38f;

  const _Float16* kbase = kT + (size_t)bh * N_ * HD_;
  const int NCH = N_ / CHUNK_;                       // 16 chunks

#if defined(HAVE_TDM)
  if (lead) {                                        // prologue: fill buffer 0
    tdm_load_chunk((unsigned)(uintptr_t)&lds[0][0], kbase);
    __builtin_amdgcn_s_wait_tensorcnt(0);
  }
  __syncthreads();
#endif

  for (int ci = 0; ci < NCH; ++ci) {
    const _Float16* buf = lds[ci & 1];
#if defined(HAVE_TDM)
    if (ci + 1 < NCH && lead)                        // async-prefetch next chunk
      tdm_load_chunk((unsigned)(uintptr_t)&lds[(ci + 1) & 1][0],
                     kbase + (size_t)(ci + 1) * CHUNK_ * HD_);
#else
    __syncthreads();
    {
      const uint4* src = (const uint4*)(kbase + (size_t)ci * CHUNK_ * HD_);
      uint4* dstl = (uint4*)lds[ci & 1];
      for (int i = threadIdx.x; i < CHUNK_ * HD_ / 8; i += 256) dstl[i] = src[i];
    }
    __syncthreads();
#endif

#pragma unroll 4
    for (int mt = 0; mt < CHUNK_; mt += 16) {
      const _Float16* brow = buf + (mt + lm) * HD_;  // [N=m][K=d], d contiguous
      v16h b0 = frag_b(brow,      lane);
      v16h b1 = frag_b(brow + 32, lane);
      v8f c;
#pragma unroll
      for (int r = 0; r < 8; ++r) c[r] = 0.f;
      c = __builtin_amdgcn_wmma_f32_16x16x32_f16(false, a0, false, b0, (short)0, c, false, false);
      c = __builtin_amdgcn_wmma_f32_16x16x32_f16(false, a1, false, b1, (short)0, c, false, false);
#pragma unroll
      for (int r = 0; r < 8; ++r) rmax[r] = fmaxf(rmax[r], c[r]);
    }

#if defined(HAVE_TDM)
    __syncthreads();                                 // all waves done with buf ci
    if (lead) __builtin_amdgcn_s_wait_tensorcnt(0);  // chunk ci+1 landed
    __syncthreads();
#endif
  }

  // lanes 0-15 hold M=r (n0+r), lanes 16-31 hold M=r+8; max across each 16-lane half
#pragma unroll
  for (int r = 0; r < 8; ++r) {
    float vv = rmax[r];
#pragma unroll
    for (int off = 8; off; off >>= 1) vv = fmaxf(vv, __shfl_xor(vv, off, 16));
    rmax[r] = vv;
  }
  if (lm == 0) {
    int nb = n0 + ((lane & 16) ? 8 : 0);
    float* dst = attnmax + (size_t)bh * N_ + nb;
#pragma unroll
    for (int r = 0; r < 8; ++r) dst[r] = rmax[r];
  }
}

// ---------------------------------------------------------------------------
// 6) out = identity + pv[b][c] * (scale * sum_h attnmax[b][h][n]) + b_proj[c]
__global__ void k_final(const float* __restrict__ identity, const float* __restrict__ pv,
                        const float* __restrict__ bproj, const float* __restrict__ attnmax,
                        float* __restrict__ out) {
  size_t i = (size_t)blockIdx.x * 256 + threadIdx.x; // B*C*N = 4M
  int n = (int)(i & (N_ - 1));
  int c = (int)((i >> 12) & (C_ - 1));
  int b = (int)(i >> 20);
  const float* am = attnmax + (size_t)b * NH_ * N_;
  float s = am[n] + am[N_ + n] + am[2 * N_ + n] + am[3 * N_ + n];
  out[i] = identity[i] + pv[b * C_ + c] * (SCALE_ * s) + bproj[c];
}

// ---------------------------------------------------------------------------
extern "C" void kernel_launch(void* const* d_in, const int* in_sizes, int n_in,
                              void* d_out, int out_size, void* d_ws, size_t ws_size,
                              hipStream_t stream) {
  const float* x        = (const float*)d_in[0];
  const float* identity = (const float*)d_in[1];
  const float* wq       = (const float*)d_in[2];
  const float* wk       = (const float*)d_in[3];
  const float* wproj    = (const float*)d_in[4];
  const float* bproj    = (const float*)d_in[5];
  float* out = (float*)d_out;

  char* ws = (char*)d_ws;
  size_t off = 0;
  _Float16* xT     = (_Float16*)(ws + off); off += (size_t)B_ * N_ * C_ * 2;      // 8 MB
  _Float16* wq16   = (_Float16*)(ws + off); off += (size_t)C_ * C_ * 2;           // 128 KB
  _Float16* wk16   = (_Float16*)(ws + off); off += (size_t)C_ * C_ * 2;           // 128 KB
  _Float16* qT     = (_Float16*)(ws + off); off += (size_t)BH_ * N_ * HD_ * 2;    // 8 MB
  _Float16* kT     = (_Float16*)(ws + off); off += (size_t)BH_ * N_ * HD_ * 2;    // 8 MB
  float*    attnmx = (float*)   (ws + off); off += (size_t)BH_ * N_ * 4;          // 256 KB
  float*    v      = (float*)   (ws + off); off += (size_t)B_ * C_ * 4;
  float*    pv     = (float*)   (ws + off); off += (size_t)B_ * C_ * 4;

  k_prep_w     <<<256,            256, 0, stream>>>(wq, wk, wq16, wk16);
  k_transpose_x<<<dim3(64, 4, 4), 256, 0, stream>>>(x, xT);
  k_vmean      <<<128,            256, 0, stream>>>(x, v);
  k_pv         <<<4,              256, 0, stream>>>(wproj, v, pv);
  k_qkproj     <<<2048,           256, 0, stream>>>(xT, wq16, wk16, qT, kT);
  k_scores     <<<512,            256, 0, stream>>>(qT, kT, attnmx);
  k_final      <<<16384,          256, 0, stream>>>(identity, pv, bproj, attnmx, out);
}
